// CausalSelfAttention_28733331211004
// MI455X (gfx1250) — matrix operand order repaired and checked by reading, not compiled, not run on hardware
//
#include <hip/hip_runtime.h>
#include <math.h>

// ---------------------------------------------------------------------------
// CDNA5 (gfx1250) causal self-attention: bf16 WMMA + TDM async-tensor staging.
// ---------------------------------------------------------------------------

typedef __attribute__((ext_vector_type(16))) __bf16 bf16x16;
typedef __attribute__((ext_vector_type(8)))  float  f32x8;
typedef __attribute__((ext_vector_type(4)))  unsigned int u32x4;
typedef __attribute__((ext_vector_type(8)))  int  i32x8;
typedef __attribute__((ext_vector_type(4)))  int  i32x4;

#define B_  4
#define T_  2048
#define C_  1024
#define H_  16
#define D_  64
#define N3_ (3 * C_)

__device__ __forceinline__ f32x8 wmma_bf16(bf16x16 a, bf16x16 b, f32x8 c) {
  // 8 args: (neg_a, A, neg_b, B, c_mod, C, reuse_a, reuse_b)
  return __builtin_amdgcn_wmma_f32_16x16x32_bf16(false, a, false, b, (short)0, c,
                                                 false, false);
}

// A-fragment: 16x32 bf16 tile, row-major source, row stride = ld elements.
// Lane l: M = l&15 ; half = l>>4 ; K = 16*half+e
__device__ __forceinline__ bf16x16 load_a_frag(const __bf16* base, int ld) {
  const int lane = threadIdx.x & 31;
  const int half = lane >> 4;
  const int m = lane & 15;
  const __bf16* p = base + (size_t)m * ld;
  bf16x16 a;
  const int k0 = half * 16;
#pragma unroll
  for (int e = 0; e < 16; ++e) a[e] = p[k0 + e];
  return a;
}

// B-fragment (32x16, entry (k,n)) from transposed storage srcT[n][k], stride ld.
// Lane: N = l&15, K = 16*(l>>4)+e  -> 16 contiguous bf16 per lane.
__device__ __forceinline__ bf16x16 load_bT_frag(const __bf16* baseT, int ld) {
  const int lane = threadIdx.x & 31;
  const int n = lane & 15;
  const __bf16* p = baseT + (size_t)n * ld + ((lane >> 4) * 16);
  bf16x16 b;
#pragma unroll
  for (int e = 0; e < 16; ++e) b[e] = p[e];
  return b;
}

// B-fragment from row-major src[k][n], stride ld (strided loads).
__device__ __forceinline__ bf16x16 load_b_frag(const __bf16* base, int ld) {
  const int lane = threadIdx.x & 31;
  const int n = lane & 15;
  const int kb = (lane >> 4) * 16;
  bf16x16 b;
#pragma unroll
  for (int e = 0; e < 16; ++e) b[e] = base[(size_t)(kb + e) * ld + n];
  return b;
}

// ---------------------- Tensor Data Mover (TDM) helper ---------------------
// 2D tile load: rows x 64 bf16 elements, tensor row stride = 64 elements
// (our K/V tensors are [T][64] bf16, row-contiguous). D# per ISA 08_async_tensor.
__device__ __forceinline__ void tdm_load_tile_rows64(const __bf16* gsrc,
                                                     void* ldst, int rows) {
  unsigned laddr = (unsigned)(unsigned long long)
      (__attribute__((address_space(3))) char*)ldst;           // LDS byte offset
  unsigned long long ga = (unsigned long long)(size_t)gsrc;

  // descriptor words are wave-uniform; pin them to SGPR-friendly values
  laddr = (unsigned)__builtin_amdgcn_readfirstlane((int)laddr);
  unsigned galo = (unsigned)__builtin_amdgcn_readfirstlane((int)(ga & 0xFFFFFFFFu));
  unsigned gahi = (unsigned)__builtin_amdgcn_readfirstlane((int)(ga >> 32));

  u32x4 g0;
  g0[0] = 1u;                                    // count=1 (valid), no gather
  g0[1] = laddr;                                 // lds_addr
  g0[2] = galo;                                  // global_addr[31:0]
  g0[3] = (gahi & 0x01FFFFFFu) | (2u << 30);     // global_addr[56:32] | type=2

  i32x8 g1;
  g1[0] = (1 << 16);                             // data_size = 1 -> 2 bytes
  g1[1] = (int)((unsigned)(D_ & 0xFFFF) << 16);  // tensor_dim0[15:0] = 64
  g1[2] = (int)(((unsigned)T_ & 0xFFFFu) << 16); // tensor_dim0 hi=0 | tensor_dim1 lo
  g1[3] = (int)((unsigned)D_ << 16);             // tensor_dim1 hi=0 | tile_dim0 = 64
  g1[4] = rows & 0xFFFF;                         // tile_dim1 = rows, tile_dim2 = 0
  g1[5] = D_;                                    // tensor_dim0_stride = 64
  g1[6] = 0;
  g1[7] = 0;

  i32x4 z4 = {0, 0, 0, 0};
#if defined(__clang_major__) && (__clang_major__ >= 23)
  i32x8 z8 = {0, 0, 0, 0, 0, 0, 0, 0};
  __builtin_amdgcn_tensor_load_to_lds(g0, g1, z4, z4, z8, 0);
#else
  __builtin_amdgcn_tensor_load_to_lds(g0, g1, z4, z4, 0);
#endif
}

// ------------------------------- converters --------------------------------

__global__ void f32_to_bf16_kernel(const float* __restrict__ in,
                                   __bf16* __restrict__ out, int n) {
  for (int i = blockIdx.x * blockDim.x + threadIdx.x; i < n;
       i += gridDim.x * blockDim.x)
    out[i] = (__bf16)in[i];
}

// w[K][N] (fp32, row-major) -> wT[N][K] (bf16)
__global__ void transpose_to_bf16_kernel(const float* __restrict__ w,
                                         __bf16* __restrict__ wT, int K, int N) {
  int total = K * N;
  for (int idx = blockIdx.x * blockDim.x + threadIdx.x; idx < total;
       idx += gridDim.x * blockDim.x) {
    int n = idx / K;
    int k = idx - n * K;
    wT[idx] = (__bf16)w[(size_t)k * N + n];
  }
}

// ------------------------------- QKV GEMM ----------------------------------
// xb [M=B*T][C] bf16 @ wT [3C][C] bf16 (+bias) -> q/k/v in [B][H][T][D] bf16.
// Wave: 16x64 strip, software-pipelined fragment loads (overlap with WMMA).
__global__ __launch_bounds__(128) void qkv_gemm_kernel(
    const __bf16* __restrict__ xb, const __bf16* __restrict__ wT,
    const float* __restrict__ bias, __bf16* __restrict__ qm,
    __bf16* __restrict__ km, __bf16* __restrict__ vm) {
  const int wave = threadIdx.x >> 5;
  const int lane = threadIdx.x & 31;
  const int row0 = blockIdx.x * 16;
  const int col0 = blockIdx.y * 256 + wave * 64;

  const __bf16* arow = xb + (size_t)row0 * C_;
  f32x8 acc[4] = {};
  bf16x16 a = load_a_frag(arow, C_);
  bf16x16 b = load_bT_frag(wT + (size_t)col0 * C_, C_);

  for (int kk = 0; kk < C_; kk += 32) {
    const int kn = (kk + 32 < C_) ? kk + 32 : 0;   // wrap to avoid OOB tail loads
    __builtin_prefetch(arow + kk + 256, 0, 0);
#pragma unroll
    for (int t = 0; t < 4; ++t) {
      bf16x16 bn = (t < 3)
          ? load_bT_frag(wT + (size_t)(col0 + (t + 1) * 16) * C_ + kk, C_)
          : load_bT_frag(wT + (size_t)col0 * C_ + kn, C_);
      bf16x16 an;
      if (t == 3) an = load_a_frag(arow + kn, C_);
      acc[t] = wmma_bf16(a, b, acc[t]);
      b = bn;
      if (t == 3) a = an;
    }
  }

  const int half = lane >> 4, nl = lane & 15;
#pragma unroll
  for (int t = 0; t < 4; ++t) {
    const int n = col0 + t * 16 + nl;        // 0..3071
    const float bv = bias[n];
    const int which = n >> 10;               // 0=q 1=k 2=v (tile never straddles)
    const int cm = n & (C_ - 1);
    const int h = cm >> 6, d = cm & (D_ - 1);
    __bf16* dst = (which == 0) ? qm : (which == 1) ? km : vm;
#pragma unroll
    for (int r = 0; r < 8; ++r) {
      const int row = row0 + r + 8 * half;
      const int bb = row >> 11;
      const int tt = row & (T_ - 1);
      dst[((((size_t)bb * H_ + h) * T_ + tt) * D_) + d] =
          (__bf16)(acc[t][r] + bv);
    }
  }
}

// ---------------------------- flash attention ------------------------------
// One wave <-> one 16-row Q tile of one (b,h). K/V 32x64 tiles staged into
// per-wave LDS double buffers by the Tensor Data Mover; online softmax.
#define FA_WAVES 2
__global__ __launch_bounds__(32 * FA_WAVES) void flash_attn_kernel(
    const __bf16* __restrict__ qm, const __bf16* __restrict__ km,
    const __bf16* __restrict__ vm, __bf16* __restrict__ y) {
  __shared__ __bf16 kvbuf[FA_WAVES][2][2][32][D_];  // [wave][buf][K|V][row][d]
  __shared__ __bf16 pbuf[FA_WAVES][16][32];         // P staging (C->A layout)

  const int wave = threadIdx.x >> 5;
  const int lane = threadIdx.x & 31;
  const int half = lane >> 4, nl = lane & 15;
  const int bh = blockIdx.y;                         // 0..B*H-1
  const int q0 = (blockIdx.x * FA_WAVES + wave) * 16;

  const __bf16* Q = qm + (size_t)bh * T_ * D_;
  const __bf16* K = km + (size_t)bh * T_ * D_;
  const __bf16* V = vm + (size_t)bh * T_ * D_;

  const bf16x16 qa0 = load_a_frag(Q + (size_t)q0 * D_ + 0, D_);
  const bf16x16 qa1 = load_a_frag(Q + (size_t)q0 * D_ + 32, D_);

  f32x8 o[4] = {};
  float m_i[8], l_i[8];
#pragma unroll
  for (int r = 0; r < 8; ++r) { m_i[r] = -3.0e38f; l_i[r] = 0.0f; }

  const float scale = 0.125f;  // 1/sqrt(64)
  const int nblk = (q0 + 15) / 32 + 1;

  // prologue: DMA block 0 K/V tiles into buffer 0
  tdm_load_tile_rows64(K, &kvbuf[wave][0][0][0][0], 32);
  tdm_load_tile_rows64(V, &kvbuf[wave][0][1][0][0], 32);

  for (int i = 0; i < nblk; ++i) {
    const int kcol0 = i * 32;
    const int cur = i & 1;
    if (i + 1 < nblk) {
      const int nk = (i + 1) * 32;
      tdm_load_tile_rows64(K + (size_t)nk * D_, &kvbuf[wave][cur ^ 1][0][0][0], 32);
      tdm_load_tile_rows64(V + (size_t)nk * D_, &kvbuf[wave][cur ^ 1][1][0][0], 32);
      __builtin_amdgcn_s_wait_tensorcnt(2);  // block i's two DMAs done (in-order)
    } else {
      __builtin_amdgcn_s_wait_tensorcnt(0);  // drain
    }
    const __bf16* Kb = &kvbuf[wave][cur][0][0][0];
    const __bf16* Vb = &kvbuf[wave][cur][1][0][0];

    // ---- S = Q @ K^T from LDS (contract over D=64) ----
    f32x8 s[2] = {};
#pragma unroll
    for (int nt = 0; nt < 2; ++nt) {
      bf16x16 b0 = load_bT_frag(Kb + (size_t)(nt * 16) * D_ + 0, D_);
      bf16x16 b1 = load_bT_frag(Kb + (size_t)(nt * 16) * D_ + 32, D_);
      s[nt] = wmma_bf16(qa0, b0, s[nt]);
      s[nt] = wmma_bf16(qa1, b1, s[nt]);
    }

    // ---- scale + causal mask + online softmax ----
#pragma unroll
    for (int r = 0; r < 8; ++r) {
      const int qrow = q0 + r + 8 * half;
      float mx = -3.0e38f;
#pragma unroll
      for (int nt = 0; nt < 2; ++nt) {
        const int kcol = kcol0 + nt * 16 + nl;
        float val = s[nt][r] * scale;
        val = (kcol <= qrow) ? val : -3.0e38f;
        s[nt][r] = val;
        mx = fmaxf(mx, val);
      }
#pragma unroll
      for (int off = 1; off < 16; off <<= 1)     // row max within half-wave
        mx = fmaxf(mx, __shfl_xor(mx, off, 16));

      const float mnew = fmaxf(m_i[r], mx);
      const float corr = __expf(m_i[r] - mnew);
      m_i[r] = mnew;
      l_i[r] *= corr;
#pragma unroll
      for (int t = 0; t < 4; ++t) o[t][r] *= corr;

      float psum = 0.0f;
#pragma unroll
      for (int nt = 0; nt < 2; ++nt) {
        const int kcol = kcol0 + nt * 16 + nl;
        const float p = (kcol <= qrow) ? __expf(s[nt][r] - mnew) : 0.0f;
        s[nt][r] = p;
        psum += p;
      }
#pragma unroll
      for (int off = 1; off < 16; off <<= 1)
        psum += __shfl_xor(psum, off, 16);
      l_i[r] += psum;
    }

    // ---- P: C-layout -> A-layout via per-wave LDS tile (DS ops in-order) ----
#pragma unroll
    for (int r = 0; r < 8; ++r) {
      const int m = r + 8 * half;
      pbuf[wave][m][nl]      = (__bf16)s[0][r];
      pbuf[wave][m][16 + nl] = (__bf16)s[1][r];
    }
    bf16x16 pa;
    {
      const int m = lane & 15;
      const int k0 = half * 16;
#pragma unroll
      for (int e = 0; e < 16; ++e) pa[e] = pbuf[wave][m][k0 + e];
    }

    // ---- O += P @ V from LDS (contract over 32 keys; N = 64 d-cols) ----
#pragma unroll
    for (int t = 0; t < 4; ++t) {
      bf16x16 bv = load_b_frag(Vb + t * 16, D_);
      o[t] = wmma_bf16(pa, bv, o[t]);
    }
  }

  // ---- normalize + write y as [B*T][C] bf16 (col = h*64 + d) ----
  const int b = bh >> 4, h = bh & 15;
#pragma unroll
  for (int r = 0; r < 8; ++r) {
    const float inv = 1.0f / l_i[r];
    const int qrow = q0 + r + 8 * half;
#pragma unroll
    for (int t = 0; t < 4; ++t) {
      const int d = t * 16 + nl;
      y[(size_t)(b * T_ + qrow) * C_ + h * D_ + d] = (__bf16)(o[t][r] * inv);
    }
  }
}

// ----------------------------- projection GEMM -----------------------------
// y [M=B*T][C] bf16 @ wprojT [C][C] bf16 + b_proj -> out fp32 [M][C]
__global__ __launch_bounds__(128) void proj_gemm_kernel(
    const __bf16* __restrict__ yb, const __bf16* __restrict__ wT,
    const float* __restrict__ bias, float* __restrict__ out) {
  const int wave = threadIdx.x >> 5;
  const int lane = threadIdx.x & 31;
  const int row0 = blockIdx.x * 16;
  const int col0 = blockIdx.y * 256 + wave * 64;

  const __bf16* arow = yb + (size_t)row0 * C_;
  f32x8 acc[4] = {};
  bf16x16 a = load_a_frag(arow, C_);
  bf16x16 b = load_bT_frag(wT + (size_t)col0 * C_, C_);

  for (int kk = 0; kk < C_; kk += 32) {
    const int kn = (kk + 32 < C_) ? kk + 32 : 0;
    __builtin_prefetch(arow + kk + 256, 0, 0);
#pragma unroll
    for (int t = 0; t < 4; ++t) {
      bf16x16 bn = (t < 3)
          ? load_bT_frag(wT + (size_t)(col0 + (t + 1) * 16) * C_ + kk, C_)
          : load_bT_frag(wT + (size_t)col0 * C_ + kn, C_);
      bf16x16 an;
      if (t == 3) an = load_a_frag(arow + kn, C_);
      acc[t] = wmma_bf16(a, b, acc[t]);
      b = bn;
      if (t == 3) a = an;
    }
  }

  const int half = lane >> 4, nl = lane & 15;
#pragma unroll
  for (int t = 0; t < 4; ++t) {
    const int n = col0 + t * 16 + nl;
    const float bv = bias[n];
#pragma unroll
    for (int r = 0; r < 8; ++r) {
      const int row = row0 + r + 8 * half;
      out[(size_t)row * C_ + n] = acc[t][r] + bv;
    }
  }
}

// --------------------------------- launch ----------------------------------

extern "C" void kernel_launch(void* const* d_in, const int* in_sizes, int n_in,
                              void* d_out, int out_size, void* d_ws,
                              size_t ws_size, hipStream_t stream) {
  (void)in_sizes; (void)n_in; (void)out_size; (void)ws_size;

  const float* x      = (const float*)d_in[0];
  const float* w_attn = (const float*)d_in[1];
  const float* b_attn = (const float*)d_in[2];
  const float* w_proj = (const float*)d_in[3];
  const float* b_proj = (const float*)d_in[4];
  float* out = (float*)d_out;

  // workspace carve-up (all bf16): ~92 MB total
  char* ws = (char*)d_ws;
  size_t off = 0;
  auto carve = [&](size_t bytes) -> void* {
    void* p = ws + off;
    off = (off + bytes + 255) & ~(size_t)255;
    return p;
  };
  const size_t M = (size_t)B_ * T_;
  __bf16* xb     = (__bf16*)carve(M * C_ * 2);            // x in bf16
  __bf16* wattnT = (__bf16*)carve((size_t)N3_ * C_ * 2);  // w_attn^T bf16
  __bf16* wprojT = (__bf16*)carve((size_t)C_ * C_ * 2);   // w_proj^T bf16
  __bf16* qm     = (__bf16*)carve(M * C_ * 2);            // [B,H,T,D]
  __bf16* km     = (__bf16*)carve(M * C_ * 2);
  __bf16* vm     = (__bf16*)carve(M * C_ * 2);
  __bf16* yb     = (__bf16*)carve(M * C_ * 2);            // attn output bf16

  f32_to_bf16_kernel<<<2048, 256, 0, stream>>>(x, xb, (int)(M * C_));
  transpose_to_bf16_kernel<<<2048, 256, 0, stream>>>(w_attn, wattnT, C_, N3_);
  transpose_to_bf16_kernel<<<1024, 256, 0, stream>>>(w_proj, wprojT, C_, C_);

  qkv_gemm_kernel<<<dim3((unsigned)(M / 16), N3_ / 256), 128, 0, stream>>>(
      xb, wattnT, b_attn, qm, km, vm);

  flash_attn_kernel<<<dim3(T_ / 16 / FA_WAVES, B_ * H_), 32 * FA_WAVES, 0,
                      stream>>>(qm, km, vm, yb);

  proj_gemm_kernel<<<dim3((unsigned)(M / 16), C_ / 256), 128, 0, stream>>>(
      yb, wprojT, b_proj, out);
}
